// GCNClassifier_11879879541074
// MI455X (gfx1250) — matrix operand order repaired and checked by reading, not compiled, not run on hardware
//
#include <hip/hip_runtime.h>
#include <hip/hip_bf16.h>

typedef __attribute__((ext_vector_type(16))) __bf16 v16bf;
typedef __attribute__((ext_vector_type(8)))  __bf16 v8bf;
typedef __attribute__((ext_vector_type(8)))  float  v8f;

#define BM 128
#define BN 128
#define BK 32

// ---------------------------------------------------------------------------
// CDNA5 async global->LDS copy (ASYNCcnt-tracked), per-lane addresses.
//   lds_addr : wave-relative LDS byte offset (low 32 bits of flat shared addr)
//   goff     : 32-bit unsigned byte offset added to 64-bit SGPR base (GVS mode)
// ---------------------------------------------------------------------------
__device__ __forceinline__ void async_b128(unsigned lds_addr, unsigned goff,
                                           const void* base) {
    asm volatile("global_load_async_to_lds_b128 %0, %1, %2"
                 :: "v"(lds_addr), "v"(goff), "s"(base) : "memory");
}
__device__ __forceinline__ void wait_async0() {
    asm volatile("s_wait_asynccnt 0" ::: "memory");
}

// ---------------------------------------------------------------------------
// Degree / normalization
// ---------------------------------------------------------------------------
__global__ __launch_bounds__(256) void k_fill1(float* __restrict__ p, int n) {
    int i = blockIdx.x * 256 + threadIdx.x;
    if (i < n) p[i] = 1.0f;  // self-loop contributes 1 to its own degree
}

__global__ __launch_bounds__(256) void k_count_deg(const long long* __restrict__ ei,
                                                   float* __restrict__ deg, int E) {
    int e = blockIdx.x * 256 + threadIdx.x;
    if (e < E) atomicAdd(&deg[(int)ei[(long long)E + e]], 1.0f);
}

__global__ __launch_bounds__(256) void k_rsqrt_inplace(float* __restrict__ p, int n) {
    int i = blockIdx.x * 256 + threadIdx.x;
    if (i < n) p[i] = 1.0f / sqrtf(p[i]);      // deg >= 1 always (self-loops)
}

// ---------------------------------------------------------------------------
// One-time W1 prep: f32 [K][N] -> bf16 hi/lo, transposed to [n][k]
// ---------------------------------------------------------------------------
__global__ __launch_bounds__(256) void k_prep_wt(const float* __restrict__ W,
                                                 __bf16* __restrict__ Whi,
                                                 __bf16* __restrict__ Wlo,
                                                 int K, int N) {
    int t = blockIdx.x * 256 + threadIdx.x;      // t = n*K + k
    if (t >= K * N) return;
    int n = t / K, k = t - n * K;
    float v = W[(size_t)k * N + n];
    __bf16 h = (__bf16)v;
    Whi[t] = h;
    Wlo[t] = (__bf16)(v - (float)h);
}

// ---------------------------------------------------------------------------
// GEMM1: H = X @ W1  (M x 768)@(768 x 256), bf16x3 split, f32 accumulate.
// A: f32 global -> convert -> packed LDS stores.  B: async global->LDS copy
// of pre-transposed bf16 hi/lo panels.  8 waves, wave tile 32x64.
// ---------------------------------------------------------------------------
__global__ __launch_bounds__(256) void k_gemm1_wmma(const float* __restrict__ X,
                                                    const __bf16* __restrict__ Whi,
                                                    const __bf16* __restrict__ Wlo,
                                                    float* __restrict__ H,
                                                    int M, int K, int N) {
    __shared__ alignas(32) __bf16 Ahi[BM][BK];
    __shared__ alignas(32) __bf16 Alo[BM][BK];
    __shared__ alignas(32) __bf16 Bhi[BN][BK];   // [n][k], copied asynchronously
    __shared__ alignas(32) __bf16 Blo[BN][BK];

    const int tid   = threadIdx.x;
    const int wave  = tid >> 5;
    const int lane  = tid & 31;
    const int l16   = lane & 15;
    const int lhalf = lane >> 4;

    const int m0 = blockIdx.y * BM;
    const int n0 = blockIdx.x * BN;
    const int wm = (wave >> 1) * 32;
    const int wn = (wave & 1) * 64;

    v8f acc[2][4];
#pragma unroll
    for (int i = 0; i < 2; ++i)
#pragma unroll
        for (int j = 0; j < 4; ++j) acc[i][j] = (v8f){0.f,0.f,0.f,0.f,0.f,0.f,0.f,0.f};

    // A staging map: 2 threads per row, 16 consecutive floats each
    const int ar = tid >> 1;
    const int ac = (tid & 1) * 16;
    // B staging map: 2 threads per n-row, 16 bf16 (32B) each
    const int brow = tid >> 1;              // 0..127
    const int bco  = (tid & 1) * 16;        // bf16 element offset in k
    const unsigned ldsBhi = (unsigned)(uintptr_t)&Bhi[brow][bco];
    const unsigned ldsBlo = (unsigned)(uintptr_t)&Blo[brow][bco];

    for (int k0 = 0; k0 < K; k0 += BK) {
        // ---- B tiles: async global->LDS (16KB total, 4x b128 per thread) ----
        {
            unsigned goff = (unsigned)(((n0 + brow) * K + k0 + bco) * 2);
            async_b128(ldsBhi,      goff,      Whi);
            async_b128(ldsBhi + 16, goff + 16, Whi);
            async_b128(ldsBlo,      goff,      Wlo);
            async_b128(ldsBlo + 16, goff + 16, Wlo);
        }
        // ---- A tile: f32 load, split to bf16 hi/lo, packed b128 stores ----
        {
            const bool valid = (m0 + ar) < M;
            const float* src = X + (size_t)(m0 + ar) * K + k0 + ac;
            if (valid && (k0 + BK) < K)
                __builtin_prefetch(src + BK, 0, 0);   // global_prefetch_b8
            float f[16];
#pragma unroll
            for (int i = 0; i < 16; i += 4) {
                float4 v = valid ? *(const float4*)(src + i) : make_float4(0.f,0.f,0.f,0.f);
                f[i] = v.x; f[i+1] = v.y; f[i+2] = v.z; f[i+3] = v.w;
            }
            union { v8bf v; __bf16 e[8]; } ph[2], pl[2];
#pragma unroll
            for (int i = 0; i < 16; ++i) {
                __bf16 h = (__bf16)f[i];
                ph[i >> 3].e[i & 7] = h;
                pl[i >> 3].e[i & 7] = (__bf16)(f[i] - (float)h);
            }
            *(v8bf*)&Ahi[ar][ac]     = ph[0].v;
            *(v8bf*)&Ahi[ar][ac + 8] = ph[1].v;
            *(v8bf*)&Alo[ar][ac]     = pl[0].v;
            *(v8bf*)&Alo[ar][ac + 8] = pl[1].v;
        }
        wait_async0();          // own-wave async copies complete in LDS
        __syncthreads();        // all waves' A stores + B copies visible

        // ---- WMMA: A-frag = 16 contiguous K per lane half (k = 16*lhalf + s) ----
        union frag { v16bf v; v8bf h[2]; };
        frag ah[2], al[2];
#pragma unroll
        for (int i = 0; i < 2; ++i) {
            const int mr = wm + i * 16 + l16;
            ah[i].h[0] = *(const v8bf*)&Ahi[mr][lhalf * 16];
            ah[i].h[1] = *(const v8bf*)&Ahi[mr][lhalf * 16 + 8];
            al[i].h[0] = *(const v8bf*)&Alo[mr][lhalf * 16];
            al[i].h[1] = *(const v8bf*)&Alo[mr][lhalf * 16 + 8];
        }
#pragma unroll
        for (int j = 0; j < 4; ++j) {
            const int nc = wn + j * 16 + l16;
            v16bf bh = *(const v16bf*)&Bhi[nc][lhalf * 16];
            v16bf bl = *(const v16bf*)&Blo[nc][lhalf * 16];
#pragma unroll
            for (int i = 0; i < 2; ++i) {
                acc[i][j] = __builtin_amdgcn_wmma_f32_16x16x32_bf16(
                    false, ah[i].v, false, bh, (short)0, acc[i][j], false, false);
                acc[i][j] = __builtin_amdgcn_wmma_f32_16x16x32_bf16(
                    false, ah[i].v, false, bl, (short)0, acc[i][j], false, false);
                acc[i][j] = __builtin_amdgcn_wmma_f32_16x16x32_bf16(
                    false, al[i].v, false, bh, (short)0, acc[i][j], false, false);
            }
        }
        __syncthreads();
    }

    // ---- store C: VGPR r -> row r (lanes 0-15) / row 8+r (lanes 16-31) ----
#pragma unroll
    for (int i = 0; i < 2; ++i) {
#pragma unroll
        for (int j = 0; j < 4; ++j) {
            const int col   = n0 + wn + j * 16 + l16;
            const int mbase = m0 + wm + i * 16 + 8 * lhalf;
#pragma unroll
            for (int r = 0; r < 8; ++r) {
                const int m = mbase + r;
                if (m < M) H[(size_t)m * N + col] = acc[i][j][r];
            }
        }
    }
}

// ---------------------------------------------------------------------------
// Scatter-add for conv layers
// ---------------------------------------------------------------------------
__global__ __launch_bounds__(256) void k_zero(float* __restrict__ p, long long n) {
    long long i = (long long)blockIdx.x * 256 + threadIdx.x;
    if (i < n) p[i] = 0.0f;
}

__global__ __launch_bounds__(256) void k_scatter_add4(const float* __restrict__ Hm,
                                                      const long long* __restrict__ ei,
                                                      const float* __restrict__ dinv,
                                                      float* __restrict__ agg,
                                                      int E, int Nn, int C) {
    const int groups = C >> 2;
    long long gt = (long long)blockIdx.x * 256 + threadIdx.x;
    long long tot = (long long)(E + Nn) * groups;
    if (gt >= tot) return;
    const int e = (int)(gt / groups);
    const int g = (int)(gt - (long long)e * groups);
    int r, c;
    if (e < E) { r = (int)ei[e]; c = (int)ei[(long long)E + e]; }
    else       { r = c = e - E; }
    const float nrm = dinv[r] * dinv[c];
    const float4 hv = *(const float4*)&Hm[(size_t)r * C + g * 4];
    float* dst = &agg[(size_t)c * C + g * 4];
    atomicAdd(dst + 0, hv.x * nrm);
    atomicAdd(dst + 1, hv.y * nrm);
    atomicAdd(dst + 2, hv.z * nrm);
    atomicAdd(dst + 3, hv.w * nrm);
}

__global__ __launch_bounds__(256) void k_bias_relu(float* __restrict__ agg,
                                                   const float* __restrict__ b,
                                                   long long n, int C) {
    long long i = (long long)blockIdx.x * 256 + threadIdx.x;
    if (i >= n) return;
    const int c = (int)(i % C);
    float v = agg[i] + b[c];
    agg[i] = v > 0.0f ? v : 0.0f;
}

// ---------------------------------------------------------------------------
// GEMM2: H2 = relu_agg @ W2  (M x 256)@(256 x 8) — tiny N, LDS-staged W2
// ---------------------------------------------------------------------------
__global__ __launch_bounds__(256) void k_gemm2(const float* __restrict__ A,
                                               const float* __restrict__ W2,
                                               float* __restrict__ H2,
                                               int M, int K, int C) {
    extern __shared__ float Ws[];
    for (int i = threadIdx.x; i < K * C; i += 256) Ws[i] = W2[i];
    __syncthreads();
    const int npb = 256 / C;
    const int n = blockIdx.x * npb + threadIdx.x / C;
    const int c = threadIdx.x % C;
    if (n >= M) return;
    const float* a = &A[(size_t)n * K];
    float s = 0.0f;
    for (int k = 0; k < K; ++k) s += a[k] * Ws[k * C + c];
    H2[(size_t)n * C + c] = s;
}

__global__ __launch_bounds__(256) void k_init_out(float* __restrict__ out,
                                                  const float* __restrict__ b,
                                                  long long n, int C) {
    long long i = (long long)blockIdx.x * 256 + threadIdx.x;
    if (i < n) out[i] = b[(int)(i % C)];
}

__global__ __launch_bounds__(256) void k_scatter_add1(const float* __restrict__ H2,
                                                      const long long* __restrict__ ei,
                                                      const float* __restrict__ dinv,
                                                      float* __restrict__ out,
                                                      int E, int Nn, int C) {
    long long gt = (long long)blockIdx.x * 256 + threadIdx.x;
    long long tot = (long long)(E + Nn) * C;
    if (gt >= tot) return;
    const int e = (int)(gt / C);
    const int c = (int)(gt - (long long)e * C);
    int r, col;
    if (e < E) { r = (int)ei[e]; col = (int)ei[(long long)E + e]; }
    else       { r = col = e - E; }
    const float nrm = dinv[r] * dinv[col];
    atomicAdd(&out[(size_t)col * C + c], H2[(size_t)r * C + c] * nrm);
}

// ---------------------------------------------------------------------------
static inline unsigned cdiv(long long a, long long b) { return (unsigned)((a + b - 1) / b); }

extern "C" void kernel_launch(void* const* d_in, const int* in_sizes, int n_in,
                              void* d_out, int out_size, void* d_ws, size_t ws_size,
                              hipStream_t stream) {
    const float*      x  = (const float*)d_in[0];
    const long long*  ei = (const long long*)d_in[1];
    const float*      W1 = (const float*)d_in[2];
    const float*      b1 = (const float*)d_in[3];
    const float*      W2 = (const float*)d_in[4];
    const float*      b2 = (const float*)d_in[5];
    float*            out = (float*)d_out;

    const int HID  = in_sizes[3];               // 256
    const int INC  = in_sizes[2] / HID;         // 768
    const int NCLS = in_sizes[5];               // 8
    const int Nn   = in_sizes[0] / INC;         // 100000
    const int E    = in_sizes[1] / 2;           // 1600000

    char* ws = (char*)d_ws;
    size_t o = 0;
    auto carve = [&](size_t bytes) { size_t r = o; o += (bytes + 255) & ~(size_t)255; return r; };
    float*  dinv = (float*)(ws + carve((size_t)Nn * 4));
    float*  H    = (float*)(ws + carve((size_t)Nn * HID * 4));
    float*  AGG  = (float*)(ws + carve((size_t)Nn * HID * 4));
    float*  H2   = (float*)(ws + carve((size_t)Nn * NCLS * 4));
    __bf16* Whi  = (__bf16*)(ws + carve((size_t)INC * HID * 2));
    __bf16* Wlo  = (__bf16*)(ws + carve((size_t)INC * HID * 2));

    // 1) degrees (with self-loops) -> dinv
    k_fill1<<<cdiv(Nn, 256), 256, 0, stream>>>(dinv, Nn);
    k_count_deg<<<cdiv(E, 256), 256, 0, stream>>>(ei, dinv, E);
    k_rsqrt_inplace<<<cdiv(Nn, 256), 256, 0, stream>>>(dinv, Nn);

    // 2) W1 -> bf16 hi/lo transposed panels; then H = X @ W1 (WMMA bf16x3)
    k_prep_wt<<<cdiv((long long)INC * HID, 256), 256, 0, stream>>>(W1, Whi, Wlo, INC, HID);
    dim3 g1(HID / BN, cdiv(Nn, BM));
    k_gemm1_wmma<<<g1, 256, 0, stream>>>(x, Whi, Wlo, H, Nn, INC, HID);

    // 3) AGG = segment_sum(norm * H[row]) ; + b1 ; relu
    const long long nh = (long long)Nn * HID;
    k_zero<<<cdiv(nh, 256), 256, 0, stream>>>(AGG, nh);
    const long long s1 = (long long)(E + Nn) * (HID / 4);
    k_scatter_add4<<<cdiv(s1, 256), 256, 0, stream>>>(H, ei, dinv, AGG, E, Nn, HID);
    k_bias_relu<<<cdiv(nh, 256), 256, 0, stream>>>(AGG, b1, nh, HID);

    // 4) H2 = AGG @ W2 ; out = b2 + segment_sum(norm * H2[row])
    k_gemm2<<<cdiv(Nn, 256 / NCLS), 256, (size_t)HID * NCLS * 4, stream>>>(AGG, W2, H2, Nn, HID, NCLS);
    const long long no = (long long)Nn * NCLS;
    k_init_out<<<cdiv(no, 256), 256, 0, stream>>>(out, b2, no, NCLS);
    const long long s2 = (long long)(E + Nn) * NCLS;
    k_scatter_add1<<<cdiv(s2, 256), 256, 0, stream>>>(H2, ei, dinv, out, E, Nn, NCLS);
}
